// Embedding_17635135717442
// MI455X (gfx1250) — matrix operand order repaired and checked by reading, not compiled, not run on hardware
//
#include <hip/hip_runtime.h>
#include <hip/hip_fp16.h>

typedef _Float16 v16h __attribute__((ext_vector_type(16)));
typedef _Float16 v8h  __attribute__((ext_vector_type(8)));
typedef float    v8f  __attribute__((ext_vector_type(8)));

// ---------------- elementwise fp32 -> fp16 ----------------
__global__ __launch_bounds__(256) void cvt_f32_f16(const float* __restrict__ in,
                                                   _Float16* __restrict__ out,
                                                   long long n) {
  long long i = (long long)blockIdx.x * blockDim.x + threadIdx.x;
  if (i < n) out[i] = (_Float16)in[i];
}

// ------- fused convert + transpose: W (blocks of KxN fp32) -> WT (blocks of NxK f16)
__global__ __launch_bounds__(256) void cvt_transpose_f32_f16(
    const float* __restrict__ W, _Float16* __restrict__ WT,
    int K, int Ncols, long long total) {
  long long i = (long long)blockIdx.x * blockDim.x + threadIdx.x;
  if (i >= total) return;
  const long long per = (long long)K * Ncols;
  const long long b = i / per;
  const long long r = i - b * per;
  const int n = (int)(r / K);
  const int k = (int)(r - (long long)n * K);
  WT[i] = (_Float16)W[b * per + (long long)k * Ncols + n];
}

// ---------------- WMMA GEMM: C[MxN] = A[MxK] * B[KxN], B given transposed -----
// One wave -> MT stacked 16x16 C tiles (shared B fragment). 4 waves = 64 cols.
// K, NC compile-time: static trip count, shift-based addressing.
// grid = (M/(16*MT), NC/64), block = 128 threads (wave32).
template <int MT, int K, int NC>
__global__ __launch_bounds__(128) void wmma_gemm_f16_t(
    const _Float16* __restrict__ A,   // M x K, row-major
    const _Float16* __restrict__ BT,  // NC x K, row-major (= B transposed)
    float* __restrict__ C) {          // M x NC, row-major
  const int lane  = threadIdx.x & 31;
  const int wave  = threadIdx.x >> 5;
  const int l15   = lane & 15;
  const int hi    = lane >> 4;
  const int tileN = blockIdx.y * 4 + wave;
  const int colB  = tileN * 16 + l15;               // B/C column held by this lane
  const int rowBase = blockIdx.x * (16 * MT);

  const _Float16* Btrow = BT + (long long)colB * K;
  const _Float16* Arow[MT];
#pragma unroll
  for (int m = 0; m < MT; ++m)
    Arow[m] = A + (long long)(rowBase + m * 16 + l15) * K;

  v8f acc[MT];
#pragma unroll
  for (int m = 0; m < MT; ++m) acc[m] = (v8f){};

#pragma unroll 4
  for (int k = 0; k < K; k += 32) {
    // B fragment: element j = B[kb+j][colB] = BT[colB][kb+j], contiguous 32B
    const int kb = k + hi * 16;
    v8h b0 = *(const v8h*)(Btrow + kb);
    v8h b1 = *(const v8h*)(Btrow + kb + 8);
    v16h bv = __builtin_shufflevector(b0, b1, 0, 1, 2, 3, 4, 5, 6, 7,
                                      8, 9, 10, 11, 12, 13, 14, 15);
    // A fragment: elements 0..15 = K[k0..k0+15], contiguous (matches B)
    const int k0 = k + hi * 16;
#pragma unroll
    for (int m = 0; m < MT; ++m) {
      v8h a0 = *(const v8h*)(Arow[m] + k0);
      v8h a1 = *(const v8h*)(Arow[m] + k0 + 8);
      v16h av = __builtin_shufflevector(a0, a1, 0, 1, 2, 3, 4, 5, 6, 7,
                                        8, 9, 10, 11, 12, 13, 14, 15);
      acc[m] = __builtin_amdgcn_wmma_f32_16x16x32_f16(
          false, av, false, bv, (short)0, acc[m], false, false);
    }
  }
  // C layout: VGPR r -> row base + r + hi*8, col = colB
#pragma unroll
  for (int m = 0; m < MT; ++m)
#pragma unroll
    for (int r = 0; r < 8; ++r)
      C[(long long)(rowBase + m * 16 + hi * 8 + r) * NC + colB] = acc[m][r];
}

// ---------------- SpMM scatter: agg[src[e]] += wv * val[e] * emb[dst[e]] ------
// Each thread handles 4 consecutive features of one edge (float4 gather from the
// L2-resident embedding table, 4 native no-return global_atomic_add_f32 scatters).
template <int COLS>
__global__ __launch_bounds__(256) void spmm_scatter(
    const int* __restrict__ src, const int* __restrict__ dst,
    const float* __restrict__ val, const float* __restrict__ wv_elem,
    const float* __restrict__ emb, float* __restrict__ agg, long long E) {
  const int tpr = COLS >> 2;  // threads per edge row
  long long gid = (long long)blockIdx.x * blockDim.x + threadIdx.x;
  long long e = gid / tpr;
  if (e >= E) return;
  const int c4 = (int)(gid - e * tpr) * 4;
  const float s = wv_elem[0] * val[e];
  const float4 v = *(const float4*)(emb + (long long)dst[e] * COLS + c4);
  float* o = agg + (long long)src[e] * COLS + c4;
  unsafeAtomicAdd(o + 0, s * v.x);
  unsafeAtomicAdd(o + 1, s * v.y);
  unsafeAtomicAdd(o + 2, s * v.z);
  unsafeAtomicAdd(o + 3, s * v.w);
}

// ---------------- epilogues ----------------
__global__ __launch_bounds__(256) void combine_relu_to_f16(
    const float* __restrict__ emb0, const float* __restrict__ agg,
    const float* __restrict__ bias, _Float16* __restrict__ out,
    int cols, long long total) {
  long long i = (long long)blockIdx.x * blockDim.x + threadIdx.x;
  if (i >= total) return;
  const int c = (int)(i % cols);
  float v = emb0[i] + 1.01f * agg[i] + bias[c];
  out[i] = (_Float16)(v > 0.0f ? v : 0.0f);
}

__global__ __launch_bounds__(256) void combine_relu_f32(
    const float* __restrict__ emb0, const float* __restrict__ agg,
    const float* __restrict__ bias, float* __restrict__ out,
    int cols, long long total) {
  long long i = (long long)blockIdx.x * blockDim.x + threadIdx.x;
  if (i >= total) return;
  const int c = (int)(i % cols);
  float v = emb0[i] + 1.01f * agg[i] + bias[c];
  out[i] = v > 0.0f ? v : 0.0f;
}

__global__ __launch_bounds__(256) void final_avg_abs(
    const float* __restrict__ y0, const float* __restrict__ y1,
    const float* __restrict__ y2, float* __restrict__ out, long long total) {
  long long i = (long long)blockIdx.x * blockDim.x + threadIdx.x;
  if (i >= total) return;
  out[i] = fabsf((y0[i] + y1[i] + y2[i]) * (1.0f / 3.0f));
}

// ---------------- host orchestration ----------------
extern "C" void kernel_launch(void* const* d_in, const int* in_sizes, int n_in,
                              void* d_out, int out_size, void* d_ws, size_t ws_size,
                              hipStream_t stream) {
  const int F = 512, H = 128, Cc = 64;
  const int N = in_sizes[0] / F;            // 20000 (multiple of 32)
  const long long E = in_sizes[3];          // 640000

  const float* x[3]   = {(const float*)d_in[0], (const float*)d_in[1], (const float*)d_in[2]};
  const int*   src[3] = {(const int*)d_in[3], (const int*)d_in[6], (const int*)d_in[9]};
  const int*   dst[3] = {(const int*)d_in[4], (const int*)d_in[7], (const int*)d_in[10]};
  const float* val[3] = {(const float*)d_in[5], (const float*)d_in[8], (const float*)d_in[11]};
  const float* W1  = (const float*)d_in[12];
  const float* wv1 = (const float*)d_in[13];
  const float* b1  = (const float*)d_in[14];
  const float* W2  = (const float*)d_in[15];
  const float* wv2 = (const float*)d_in[16];
  const float* b2  = (const float*)d_in[17];

  // carve workspace (~135 MB)
  char* ws = (char*)d_ws;
  size_t off = 0;
  auto carve = [&](size_t bytes) -> char* {
    char* p = ws + off;
    off += (bytes + 255) & ~(size_t)255;
    return p;
  };
  _Float16* xh[3];
  for (int i = 0; i < 3; ++i) xh[i] = (_Float16*)carve((size_t)N * F * 2);
  _Float16* w1t = (_Float16*)carve((size_t)9 * F * H * 2);   // 9 blocks of H x F (B^T)
  _Float16* w2t = (_Float16*)carve((size_t)9 * H * Cc * 2);  // 9 blocks of Cc x H
  _Float16* hh[3];
  for (int i = 0; i < 3; ++i) hh[i] = (_Float16*)carve((size_t)N * H * 2);
  float* emb[3];
  for (int i = 0; i < 3; ++i) emb[i] = (float*)carve((size_t)N * H * 4);
  float* agg = (float*)carve((size_t)N * H * 4);
  float* y[3];
  for (int i = 0; i < 3; ++i) y[i] = (float*)carve((size_t)N * Cc * 4);
  (void)ws_size; (void)n_in; (void)out_size;

  const int CT = 256;
  auto blocks = [&](long long n) { return (unsigned)((n + CT - 1) / CT); };

  // one-time conversions: x -> f16, W -> f16 transposed (B^T, row = out-channel)
  const long long nx = (long long)N * F;
  for (int i = 0; i < 3; ++i)
    cvt_f32_f16<<<blocks(nx), CT, 0, stream>>>(x[i], xh[i], nx);
  cvt_transpose_f32_f16<<<blocks(9LL * F * H), CT, 0, stream>>>(W1, w1t, F, H, 9LL * F * H);
  cvt_transpose_f32_f16<<<blocks(9LL * H * Cc), CT, 0, stream>>>(W2, w2t, H, Cc, 9LL * H * Cc);

  // view permutations: h1 [x1,x2,x3]/[a1,a2,a3]; h2 [x2,x1,x3]/[a2,a1,a3]; h3 [x3,x1,x2]/[a3,a1,a2]
  const int perm[3][3] = {{0, 1, 2}, {1, 0, 2}, {2, 0, 1}};
  const int MT = 2;                 // M-tiles per wave
  const dim3 gB(128);

  // -------- layer 1: F -> H --------
  const long long tH = (long long)N * H;
  for (int g = 0; g < 3; ++g) {
    hipMemsetAsync(agg, 0, (size_t)N * H * 4, stream);
    for (int i = 0; i < 3; ++i)
      wmma_gemm_f16_t<2, 512, 128><<<dim3(N / (16 * MT), H / 64), gB, 0, stream>>>(
          xh[perm[g][i]], w1t + (long long)(g * 3 + i) * F * H, emb[i]);
    for (int i = 0; i < 3; ++i) {
      const int p = perm[g][i];
      const long long tot = E * (H / 4);
      spmm_scatter<128><<<blocks(tot), CT, 0, stream>>>(
          src[p], dst[p], val[p], wv1 + (g * 3 + i), emb[i], agg, E);
    }
    combine_relu_to_f16<<<blocks(tH), CT, 0, stream>>>(
        emb[0], agg, b1 + g * H, hh[g], H, tH);
  }

  // -------- layer 2: H -> C --------
  const long long tC = (long long)N * Cc;
  for (int g = 0; g < 3; ++g) {
    hipMemsetAsync(agg, 0, (size_t)N * Cc * 4, stream);
    for (int i = 0; i < 3; ++i)
      wmma_gemm_f16_t<2, 128, 64><<<dim3(N / (16 * MT), Cc / 64), gB, 0, stream>>>(
          hh[perm[g][i]], w2t + (long long)(g * 3 + i) * H * Cc, emb[i]);
    for (int i = 0; i < 3; ++i) {
      const int p = perm[g][i];
      const long long tot = E * (Cc / 4);
      spmm_scatter<64><<<blocks(tot), CT, 0, stream>>>(
          src[p], dst[p], val[p], wv2 + (g * 3 + i), emb[i], agg, E);
    }
    combine_relu_f32<<<blocks(tC), CT, 0, stream>>>(
        emb[0], agg, b2 + g * Cc, y[g], Cc, tC);
  }

  // -------- final: abs(mean) --------
  final_avg_abs<<<blocks(tC), CT, 0, stream>>>(y[0], y[1], y[2], (float*)d_out, tC);
}
